// MultiHeadedAttentionLM_30322469110487
// MI455X (gfx1250) — matrix operand order repaired and checked by reading, not compiled, not run on hardware
//
#include <hip/hip_runtime.h>

// ---------------------------------------------------------------------------
// MI455X (gfx1250) GPT-block forward. wave32, WMMA bf16 16x16x32, f32 accum.
// ---------------------------------------------------------------------------

typedef unsigned short u16;
typedef __attribute__((ext_vector_type(16))) __bf16 v16bf;
typedef __attribute__((ext_vector_type(8)))  float  v8f;

union FragU {            // 16 bf16 operand fragment (8 VGPRs)
    v16bf v;
    u16   u[16];
    uint4 q[2];
};
union AccU {             // 16x16 f32 accumulator (8 VGPRs)
    v8f   v;
    float f[8];
};

__device__ __forceinline__ u16 f2bf(float f) {
    union { float f; unsigned u; } x; x.f = f;
    unsigned r = x.u + 0x7FFFu + ((x.u >> 16) & 1u);   // round-to-nearest-even
    return (u16)(r >> 16);
}

#define WMMA_BF16(A, B, C) \
    __builtin_amdgcn_wmma_f32_16x16x32_bf16(false, (A), false, (B), (short)0, (C), false, false)

// ---------------------------------------------------------------------------
// x[b,t,:] = tok_emb[ids[b,t],:] + pos_emb[t,:]   -> bf16
// ---------------------------------------------------------------------------
__global__ __launch_bounds__(256)
void embed_kernel(const int* __restrict__ ids, const float* __restrict__ tok,
                  const float* __restrict__ pos, u16* __restrict__ xbf,
                  int BT, int Dd, int T) {
    long i = (long)blockIdx.x * blockDim.x + threadIdx.x;
    if (i >= (long)BT * Dd) return;
    long bt = i / Dd;
    int  d  = (int)(i - bt * Dd);
    int  t  = (int)(bt % T);
    float v = tok[(long)ids[bt] * Dd + d] + pos[(long)t * Dd + d];
    xbf[i] = f2bf(v);
}

// ---------------------------------------------------------------------------
// out[z][c][r] = bf16(in[z][r][c])   (weight convert + transpose to N-major)
// ---------------------------------------------------------------------------
__global__ __launch_bounds__(256)
void convert_transpose_kernel(const float* __restrict__ in, u16* __restrict__ out,
                              int R, int C) {
    long total = (long)R * C;
    long i = (long)blockIdx.x * blockDim.x + threadIdx.x;
    if (i >= total) return;
    long base = (long)blockIdx.z * total;
    int r = (int)(i / C);
    int c = (int)(i - (long)r * C);
    out[base + (long)c * R + r] = f2bf(in[base + i]);
}

// ---------------------------------------------------------------------------
// Batched WMMA GEMM:  C[z] = A[z] (MxK, row-major) * B[z] (KxN, stored N-major
// as B[n*ldb+k]) (+bias).  Template:
//   OUT_BF16 : output bf16 (else f32, nontemporal stores)
//   C_T      : store C transposed: C[n*ldc + m]
// batch z -> (i = z/bz, j = z%bz); operand offsets i*s?0 + j*s?1 (elements).
// Requires: K%64==0, M%64==0. N ragged OK (addresses clamped, stores guarded).
// Block = 128 threads = 4 waves in 2x2; wave tile 32x32 (2x2 WMMA tiles);
// block tile 64x64. K-loop is double-buffered (software pipelined).
// ---------------------------------------------------------------------------
template <bool OUT_BF16, bool C_T>
__global__ __launch_bounds__(128)
void wmma_gemm_kernel(const u16* __restrict__ A, const u16* __restrict__ Bm,
                      char* __restrict__ C, const float* __restrict__ bias,
                      int M, int N, int Kin, int lda, int ldb, int ldc,
                      int bz, long sA0, long sA1, long sB0, long sB1,
                      long sC0, long sC1) {
    const int z = blockIdx.z;
    const int bi = z / bz, bj = z - bi * bz;
    const u16* Ab = A + bi * sA0 + bj * sA1;
    const u16* Bb = Bm + bi * sB0 + bj * sB1;
    const long cOff = bi * sC0 + bj * sC1;

    const int wave = threadIdx.x >> 5;
    const int lane = threadIdx.x & 31;
    const int half = lane >> 4;
    const int l16  = lane & 15;
    const int wr   = wave >> 1;          // wave row (0..1)
    const int wc   = wave & 1;           // wave col (0..1)

    const int m0 = blockIdx.y * 64 + wr * 32;   // wave's 32 rows
    const int n0 = blockIdx.x * 64 + wc * 32;   // wave's 32 cols
    if (m0 >= M) return;

    const int rowA0 = m0 + l16;
    const int rowA1 = rowA0 + 16;
    const int cB0 = (n0 + l16      < N) ? (n0 + l16)      : (N - 1);
    const int cB1 = (n0 + 16 + l16 < N) ? (n0 + 16 + l16) : (N - 1);

    const u16* a0p = Ab + (long)rowA0 * lda + half * 8;
    const u16* a1p = Ab + (long)rowA1 * lda + half * 8;
    const u16* b0p = Bb + (long)cB0 * ldb + half * 8;
    const u16* b1p = Bb + (long)cB1 * ldb + half * 8;

    AccU acc[2][2];
#pragma unroll
    for (int mi = 0; mi < 2; ++mi)
#pragma unroll
        for (int ni = 0; ni < 2; ++ni)
#pragma unroll
            for (int r = 0; r < 8; ++r) acc[mi][ni].f[r] = 0.f;

    // Two explicit fragment sets for double buffering (no dynamic indexing).
    FragU xa0, xa1, xb0, xb1;   // set X
    FragU ya0, ya1, yb0, yb1;   // set Y

#define LOAD_SET(s_a0, s_a1, s_b0, s_b1, K0)                                  \
    do {                                                                      \
        s_a0.q[0] = *(const uint4*)(a0p + (K0));                              \
        s_a0.q[1] = *(const uint4*)(a0p + (K0) + 16);                         \
        s_a1.q[0] = *(const uint4*)(a1p + (K0));                              \
        s_a1.q[1] = *(const uint4*)(a1p + (K0) + 16);                         \
        s_b0.q[0] = *(const uint4*)(b0p + (K0));                              \
        s_b0.q[1] = *(const uint4*)(b0p + (K0) + 16);                         \
        s_b1.q[0] = *(const uint4*)(b1p + (K0));                              \
        s_b1.q[1] = *(const uint4*)(b1p + (K0) + 16);                         \
    } while (0)

#define MMA_SET(s_a0, s_a1, s_b0, s_b1)                                       \
    do {                                                                      \
        acc[0][0].v = WMMA_BF16(s_a0.v, s_b0.v, acc[0][0].v);                 \
        acc[0][1].v = WMMA_BF16(s_a0.v, s_b1.v, acc[0][1].v);                 \
        acc[1][0].v = WMMA_BF16(s_a1.v, s_b0.v, acc[1][0].v);                 \
        acc[1][1].v = WMMA_BF16(s_a1.v, s_b1.v, acc[1][1].v);                 \
    } while (0)

    // Software-pipelined over pairs of 32-wide k-tiles (Kin % 64 == 0).
    LOAD_SET(xa0, xa1, xb0, xb1, 0);
    int k0 = 0;
    for (; k0 < Kin - 64; k0 += 64) {
        LOAD_SET(ya0, ya1, yb0, yb1, k0 + 32);
        MMA_SET(xa0, xa1, xb0, xb1);
        LOAD_SET(xa0, xa1, xb0, xb1, k0 + 64);
        MMA_SET(ya0, ya1, yb0, yb1);
    }
    LOAD_SET(ya0, ya1, yb0, yb1, k0 + 32);
    MMA_SET(xa0, xa1, xb0, xb1);
    MMA_SET(ya0, ya1, yb0, yb1);

#undef LOAD_SET
#undef MMA_SET

#pragma unroll
    for (int ni = 0; ni < 2; ++ni) {
        const int cg = n0 + ni * 16 + l16;
        if (cg >= N) continue;
        const float bv = bias ? bias[cg] : 0.f;
#pragma unroll
        for (int mi = 0; mi < 2; ++mi) {
#pragma unroll
            for (int r = 0; r < 8; ++r) {
                const int rg = m0 + mi * 16 + half * 8 + r;
                if (rg >= M) continue;
                const float val = acc[mi][ni].f[r] + bv;
                const long off = C_T ? ((long)cg * ldc + rg)
                                     : ((long)rg * ldc + cg);
                if (OUT_BF16) {
                    ((u16*)C)[cOff + off] = f2bf(val);
                } else {
                    __builtin_nontemporal_store(val, (float*)C + cOff + off);
                }
            }
        }
    }
}

// ---------------------------------------------------------------------------
// Fused scores + causal mask + softmax.
// grid = (T/16, H*B), block = 128 (4 waves). Each block: 16 query rows,
// full T-key strip in LDS. q,k layout (H*B, T, Kh). attn out (H*B, T, T) bf16.
// ---------------------------------------------------------------------------
__global__ __launch_bounds__(128)
void attn_softmax_kernel(const u16* __restrict__ q, const u16* __restrict__ kmat,
                         u16* __restrict__ attn, int T, int Kh) {
    __shared__ float sc[16][1032];           // 16 rows x T scores (padded)
    __shared__ float red[16][8];

    const int bh  = blockIdx.y;
    const int m0  = blockIdx.x * 16;
    const int tid = threadIdx.x;
    const int wave = tid >> 5, lane = tid & 31;
    const int half = lane >> 4, l16 = lane & 15;

    const u16* qb = q    + (long)bh * T * Kh;
    const u16* kb = kmat + (long)bh * T * Kh;

    // Q fragments are invariant across key tiles: load once (Kh=64 -> 2 frags)
    FragU fa0, fa1;
    {
        const u16* ap = qb + (long)(m0 + l16) * Kh + half * 8;
        fa0.q[0] = *(const uint4*)(ap);
        fa0.q[1] = *(const uint4*)(ap + 16);
        fa1.q[0] = *(const uint4*)(ap + 32);
        fa1.q[1] = *(const uint4*)(ap + 48);
    }

    for (int st = wave; st < (T >> 4); st += 4) {
        const int s0 = st * 16;
        AccU acc;
#pragma unroll
        for (int r = 0; r < 8; ++r) acc.f[r] = 0.f;
        const u16* bp = kb + (long)(s0 + l16) * Kh + half * 8;
        FragU fb0, fb1;
        fb0.q[0] = *(const uint4*)(bp);
        fb0.q[1] = *(const uint4*)(bp + 16);
        fb1.q[0] = *(const uint4*)(bp + 32);
        fb1.q[1] = *(const uint4*)(bp + 48);
        acc.v = WMMA_BF16(fa0.v, fb0.v, acc.v);
        acc.v = WMMA_BF16(fa1.v, fb1.v, acc.v);
#pragma unroll
        for (int r = 0; r < 8; ++r) {
            const int rl = half * 8 + r;
            const int cg = s0 + l16;
            const float v = acc.f[r] * 0.125f;      // 1/sqrt(64)
            sc[rl][cg] = (cg <= m0 + rl) ? v : -__builtin_inff();
        }
    }
    __syncthreads();

    // softmax: 8 threads per row
    const int row = tid >> 3, sub = tid & 7;
    float mx = -__builtin_inff();
    for (int c = sub; c < T; c += 8) mx = fmaxf(mx, sc[row][c]);
    red[row][sub] = mx;
    __syncthreads();
    float rmax = red[row][0];
#pragma unroll
    for (int i2 = 1; i2 < 8; ++i2) rmax = fmaxf(rmax, red[row][i2]);
    __syncthreads();

    float s = 0.f;
    for (int c = sub; c < T; c += 8) {
        const float e = __expf(sc[row][c] - rmax);
        sc[row][c] = e;
        s += e;
    }
    red[row][sub] = s;
    __syncthreads();
    float rsum = 0.f;
#pragma unroll
    for (int i2 = 0; i2 < 8; ++i2) rsum += red[row][i2];
    const float inv = 1.f / rsum;

    u16* ab = attn + ((long)bh * T + (m0 + row)) * T;
    for (int c = sub; c < T; c += 8) ab[c] = f2bf(sc[row][c] * inv);
}

// ---------------------------------------------------------------------------
// Host side
// ---------------------------------------------------------------------------
extern "C" void kernel_launch(void* const* d_in, const int* in_sizes, int n_in,
                              void* d_out, int out_size, void* d_ws, size_t ws_size,
                              hipStream_t stream) {
    (void)in_sizes; (void)n_in; (void)out_size; (void)ws_size;

    constexpr int  Bn = 4, T = 1024, D = 1024, H = 16, KH = 64, V = 50257;
    constexpr long BT = (long)Bn * T;      // 4096
    constexpr long HD = (long)H * D;       // 16384

    const int*   token_ids = (const int*)  d_in[0];
    const float* tok_emb   = (const float*)d_in[1];
    const float* pos_emb   = (const float*)d_in[2];
    const float* Wq        = (const float*)d_in[3];
    const float* Wk        = (const float*)d_in[4];
    const float* Wv        = (const float*)d_in[5];
    const float* W_proj    = (const float*)d_in[6];
    const float* b_proj    = (const float*)d_in[7];
    const float* W_out     = (const float*)d_in[8];
    const float* b_out     = (const float*)d_in[9];
    float*       logits    = (float*)d_out;

    // ---- workspace carve-out (bf16 buffers, 256B aligned) ----
    char*  ws  = (char*)d_ws;
    size_t off = 0;
    auto take = [&](size_t elems) {
        size_t o = off;
        off += ((elems * 2 + 255) & ~(size_t)255);
        return o;
    };
    u16* xbf  = (u16*)(ws + take((size_t)BT * D));        // (BT, D)
    u16* qd   = (u16*)(ws + take((size_t)H * BT * KH));   // (H*B, T, KH)
    u16* kd   = (u16*)(ws + take((size_t)H * BT * KH));   // (H*B, T, KH)
    u16* vT   = (u16*)(ws + take((size_t)H * D * BT));    // (H, D, B*T) transposed
    u16* attn = (u16*)(ws + take((size_t)H * Bn * T * T));// (H*B, T, T)
    u16* outc = (u16*)(ws + take((size_t)BT * HD));       // (BT, H*D) concat
    u16* hid  = (u16*)(ws + take((size_t)BT * D));        // (BT, D)
    u16* wqT  = (u16*)(ws + take((size_t)H * KH * D));    // (H, KH, D)
    u16* wkT  = (u16*)(ws + take((size_t)H * KH * D));
    u16* wvT  = (u16*)(ws + take((size_t)H * D * D));     // (H, D, D)
    u16* wpT  = (u16*)(ws + take((size_t)D * HD));        // (D, HD)
    u16* woT  = (u16*)(ws + take((size_t)V * D));         // (V, D)

    // ---- 1) embedding gather + bf16 ----
    {
        long n = BT * D;
        embed_kernel<<<dim3((unsigned)((n + 255) / 256)), 256, 0, stream>>>(
            token_ids, tok_emb, pos_emb, xbf, (int)BT, D, T);
    }
    // ---- 2) weight convert + transpose to N-major bf16 ----
    {
        long n = (long)D * KH;  // per head
        convert_transpose_kernel<<<dim3((unsigned)((n + 255) / 256), 1, H), 256, 0, stream>>>(Wq, wqT, D, KH);
        convert_transpose_kernel<<<dim3((unsigned)((n + 255) / 256), 1, H), 256, 0, stream>>>(Wk, wkT, D, KH);
        long nv = (long)D * D;
        convert_transpose_kernel<<<dim3((unsigned)((nv + 255) / 256), 1, H), 256, 0, stream>>>(Wv, wvT, D, D);
        long np = HD * D;
        convert_transpose_kernel<<<dim3((unsigned)((np + 255) / 256), 1, 1), 256, 0, stream>>>(W_proj, wpT, (int)HD, D);
        long no = (long)D * V;
        convert_transpose_kernel<<<dim3((unsigned)((no + 255) / 256), 1, 1), 256, 0, stream>>>(W_out, woT, D, V);
    }

    // ---- 3) Q = x @ Wq[h]   (batch h; M=BT, N=KH, K=D) ----
    wmma_gemm_kernel<true, false><<<dim3(1, (unsigned)(BT / 64), H), 128, 0, stream>>>(
        xbf, wqT, (char*)qd, nullptr,
        (int)BT, KH, D, D, D, KH,
        1, 0L, 0L, (long)KH * D, 0L, (long)BT * KH, 0L);
    // ---- 4) K = x @ Wk[h] ----
    wmma_gemm_kernel<true, false><<<dim3(1, (unsigned)(BT / 64), H), 128, 0, stream>>>(
        xbf, wkT, (char*)kd, nullptr,
        (int)BT, KH, D, D, D, KH,
        1, 0L, 0L, (long)KH * D, 0L, (long)BT * KH, 0L);
    // ---- 5) V = x @ Wv[h], stored transposed as (H, D, B*T) ----
    wmma_gemm_kernel<true, true><<<dim3(D / 64, (unsigned)(BT / 64), H), 128, 0, stream>>>(
        xbf, wvT, (char*)vT, nullptr,
        (int)BT, D, D, D, D, (int)BT,
        1, 0L, 0L, (long)D * D, 0L, (long)D * BT, 0L);

    // ---- 6) causal softmax(q kT / 8) -> attn bf16 ----
    attn_softmax_kernel<<<dim3(T / 16, H * Bn), 128, 0, stream>>>(qd, kd, attn, T, KH);

    // ---- 7) out[b,t,h*D+e] = attn[h,b] @ v[h,b]  (batch z=h*B+b) ----
    wmma_gemm_kernel<true, false><<<dim3(D / 64, T / 64, H * Bn), 128, 0, stream>>>(
        attn, vT, (char*)outc, nullptr,
        T, D, T, T, (int)BT, (int)HD,
        Bn, (long)Bn * T * T, (long)T * T,       // A: h, b strides
            (long)D * BT,     (long)T,           // B: vT (H,D,B*T); +b*T column base
            (long)D,          (long)T * HD);     // C: +h*D, +b*T*HD

    // ---- 8) hidden = out @ W_proj + b_proj (bf16) ----
    wmma_gemm_kernel<true, false><<<dim3(D / 64, (unsigned)(BT / 64), 1), 128, 0, stream>>>(
        outc, wpT, (char*)hid, b_proj,
        (int)BT, D, (int)HD, (int)HD, (int)HD, D,
        1, 0L, 0L, 0L, 0L, 0L, 0L);

    // ---- 9) logits = hidden @ W_out + b_out (f32 NT stores, ragged N) ----
    wmma_gemm_kernel<false, false><<<dim3((V + 63) / 64, (unsigned)(BT / 64), 1), 128, 0, stream>>>(
        hid, woT, (char*)logits, b_out,
        (int)BT, V, D, D, D, V,
        1, 0L, 0L, 0L, 0L, 0L, 0L);
}
